// MultiHeadAttention_25769803911
// MI455X (gfx1250) — compile-verified, matrix operand order checked by reading, not run on hardware
//
#include <hip/hip_runtime.h>
#include <hip/hip_bf16.h>

// ---------------------------------------------------------------------------
// Types for CDNA5 WMMA
// ---------------------------------------------------------------------------
typedef __attribute__((ext_vector_type(16))) __bf16 v16bf;
typedef __attribute__((ext_vector_type(8)))  float  v8f;

#define B_SZ     2
#define P_SZ     2048
#define M_SZ     1024
#define H_SZ     16
#define D_SZ     64

// f32 -> bf16 with round-to-nearest-even
__device__ __forceinline__ unsigned short f2bf(float f) {
    unsigned int u = __float_as_uint(f);
    u += 0x7FFFu + ((u >> 16) & 1u);
    return (unsigned short)(u >> 16);
}

// ---------------------------------------------------------------------------
// Load a 16x32 bf16 fragment in the CDNA5 16-bit A-matrix layout.
//   lane L (wave32): row = L&15, half h = L>>4
//   VGPR0..3 = K: h*8 .. h*8+7        (16 contiguous bytes)
//   VGPR4..7 = K: 16+h*8 .. 16+h*8+7  (16 contiguous bytes)
// Works for the B operand too when the source matrix is stored transposed
// ([N][K], rows = N), since B mirrors A with M<->N.
// `stride` = elements between consecutive rows (multiple of 8).
// ---------------------------------------------------------------------------
__device__ __forceinline__ v16bf load_frag16(const unsigned short* base, int stride) {
    const int lane = threadIdx.x & 31;
    const int r = lane & 15;
    const int h = lane >> 4;
    const unsigned short* p = base + (size_t)r * (size_t)stride + h * 8;
    union { uint4 q[2]; v16bf f; } u;
    u.q[0] = *(const uint4*)(p);
    u.q[1] = *(const uint4*)(p + 16);
    return u.f;
}

__device__ __forceinline__ v8f wmma_bf16(v16bf a, v16bf b, v8f c) {
    return __builtin_amdgcn_wmma_f32_16x16x32_bf16(
        /*neg_a=*/false, a, /*neg_b=*/false, b,
        /*c_mod=*/(short)0, c, /*reuse_a=*/false, /*reuse_b=*/false);
}

// ---------------------------------------------------------------------------
// Prep kernels: cast / transpose to bf16
// ---------------------------------------------------------------------------
__global__ void cast_f32_bf16_kernel(const float* __restrict__ in,
                                     unsigned short* __restrict__ out, int n) {
    for (int i = blockIdx.x * blockDim.x + threadIdx.x; i < n;
         i += gridDim.x * blockDim.x)
        out[i] = f2bf(in[i]);
}

// W: [H][M][D] f32  ->  Wt: [H][D][M] bf16
__global__ void transpose_whead_kernel(const float* __restrict__ W,
                                       unsigned short* __restrict__ Wt) {
    int idx = blockIdx.x * blockDim.x + threadIdx.x;           // H*M*D threads
    int h = idx / (M_SZ * D_SZ);
    int rem = idx - h * (M_SZ * D_SZ);
    int m = rem / D_SZ;
    int d = rem - m * D_SZ;
    Wt[((size_t)h * D_SZ + d) * M_SZ + m] = f2bf(W[idx]);
}

// W: [M][N] f32 -> Wt: [N][M] bf16
__global__ void transpose_wout_kernel(const float* __restrict__ W,
                                      unsigned short* __restrict__ Wt) {
    int idx = blockIdx.x * blockDim.x + threadIdx.x;           // M*N threads
    int m = idx / M_SZ;
    int n = idx - m * M_SZ;
    Wt[(size_t)n * M_SZ + m] = f2bf(W[idx]);
}

// ---------------------------------------------------------------------------
// QKV projection:  out[b][h][p][d] = X[b][p][:] . W[h][:][d]
//   Xbf : [B][P][M] bf16
//   Wt  : [H][D][M] bf16 (transposed weights)
//   Out : transposeOut==0 -> [B][H][P][D]   (Q, K)
//         transposeOut==1 -> [B][H][D][P]   (V, pre-transposed for PV matmul)
// Block = 256 threads = 8 waves; each wave computes a 16x64 tile.
// ---------------------------------------------------------------------------
__global__ __launch_bounds__(256) void qkv_proj_kernel(
    const unsigned short* __restrict__ Xbf,
    const unsigned short* __restrict__ Wt,
    unsigned short* __restrict__ Out,
    int transposeOut) {
    const int mTile = blockIdx.x;                 // 0..15 (tiles of 128 rows)
    const int head  = blockIdx.y;                 // 0..15
    const int batch = blockIdx.z;                 // 0..1
    const int wave  = threadIdx.x >> 5;
    const int lane  = threadIdx.x & 31;
    const int q0 = mTile * 128 + wave * 16;

    const unsigned short* Arow  = Xbf + ((size_t)batch * P_SZ + q0) * M_SZ;
    const unsigned short* Bbase = Wt  + (size_t)head * D_SZ * M_SZ;

    v8f acc[4] = {};
    for (int k = 0; k < M_SZ; k += 32) {
        v16bf a = load_frag16(Arow + k, M_SZ);
#pragma unroll
        for (int n = 0; n < 4; ++n) {
            v16bf b = load_frag16(Bbase + (size_t)(n * 16) * M_SZ + k, M_SZ);
            acc[n] = wmma_bf16(a, b, acc[n]);
        }
    }

    const int r  = lane & 15;
    const int hh = lane >> 4;
    const size_t bh = (size_t)batch * H_SZ + head;
#pragma unroll
    for (int n = 0; n < 4; ++n)
#pragma unroll
        for (int j = 0; j < 8; ++j) {
            int row = q0 + j + 8 * hh;            // token index
            int col = n * 16 + r;                 // d_head index
            unsigned short v = f2bf(acc[n][j]);
            if (!transposeOut)
                Out[(bh * P_SZ + row) * D_SZ + col] = v;
            else
                Out[(bh * D_SZ + col) * P_SZ + row] = v;
        }
}

// ---------------------------------------------------------------------------
// Flash attention (causal, online softmax).
//   Qb : [B][H][P][D] bf16, Kb : [B][H][P][D] bf16, Vt : [B][H][D][P] bf16
//   AttnC : [B][P][H*D] bf16 (heads concatenated, ready for out projection)
// Block = 8 waves; each wave owns 16 query rows, streams src tiles of 32.
// ---------------------------------------------------------------------------
__global__ __launch_bounds__(256) void flash_attn_kernel(
    const unsigned short* __restrict__ Qb,
    const unsigned short* __restrict__ Kb,
    const unsigned short* __restrict__ Vt,
    unsigned short* __restrict__ AttnC) {
    __shared__ unsigned short ldsP[8 * 16 * 32];   // 1KB per wave P-staging

    const int qTile = blockIdx.x;                  // 0..15
    const int head  = blockIdx.y;
    const int batch = blockIdx.z;
    const int wave  = threadIdx.x >> 5;
    const int lane  = threadIdx.x & 31;
    const int q0 = qTile * 128 + wave * 16;
    const size_t bh = (size_t)batch * H_SZ + head;

    // Q fragments for this wave's 16 rows (K = 0..31, 32..63) stay in VGPRs
    const unsigned short* Qrow = Qb + (bh * P_SZ + q0) * D_SZ;
    const v16bf qa0 = load_frag16(Qrow + 0,  D_SZ);
    const v16bf qa1 = load_frag16(Qrow + 32, D_SZ);

    v8f   o[4] = {};
    float mrow[8], lrow[8];
#pragma unroll
    for (int j = 0; j < 8; ++j) { mrow[j] = -__builtin_inff(); lrow[j] = 0.f; }

    unsigned short* myP = ldsP + wave * 16 * 32;
    const int r  = lane & 15;
    const int hh = lane >> 4;

    const int srcEnd = q0 + 16;                    // causal bound (exclusive)
    for (int s0 = 0; s0 < srcEnd; s0 += 32) {
        // ---- scores: S = Q . K^T, two 16x16 tiles (src cols s0..s0+31) ----
        v8f sc[2] = {};
#pragma unroll
        for (int t = 0; t < 2; ++t) {
            const unsigned short* Krow = Kb + (bh * P_SZ + s0 + t * 16) * D_SZ;
            v16bf kb0 = load_frag16(Krow + 0,  D_SZ);
            v16bf kb1 = load_frag16(Krow + 32, D_SZ);
            sc[t] = wmma_bf16(qa0, kb0, sc[t]);
            sc[t] = wmma_bf16(qa1, kb1, sc[t]);
        }

        // ---- scale, causal mask, online row max (16-lane reductions) ----
        float newm[8];
#pragma unroll
        for (int j = 0; j < 8; ++j) {
            const int qAbs = q0 + j + 8 * hh;
#pragma unroll
            for (int t = 0; t < 2; ++t) {
                const int sAbs = s0 + t * 16 + r;
                float v = sc[t][j] * 0.125f;       // 1/sqrt(64)
                sc[t][j] = (sAbs <= qAbs) ? v : -__builtin_inff();
            }
            float mx = fmaxf(sc[0][j], sc[1][j]);
            mx = fmaxf(mx, __shfl_xor(mx, 1, 32));
            mx = fmaxf(mx, __shfl_xor(mx, 2, 32));
            mx = fmaxf(mx, __shfl_xor(mx, 4, 32));
            mx = fmaxf(mx, __shfl_xor(mx, 8, 32));
            newm[j] = fmaxf(mrow[j], mx);
        }

        // ---- exp, row-sum, stage P into LDS, rescale O ----
#pragma unroll
        for (int j = 0; j < 8; ++j) {
            float rs = 0.f;
#pragma unroll
            for (int t = 0; t < 2; ++t) {
                float e = __expf(sc[t][j] - newm[j]);   // exp(-inf)=0 masks
                rs += e;
                myP[(j + 8 * hh) * 32 + t * 16 + r] = f2bf(e);
            }
            rs += __shfl_xor(rs, 1, 32);
            rs += __shfl_xor(rs, 2, 32);
            rs += __shfl_xor(rs, 4, 32);
            rs += __shfl_xor(rs, 8, 32);
            const float scale = __expf(mrow[j] - newm[j]);
            lrow[j] = lrow[j] * scale + rs;
            mrow[j] = newm[j];
#pragma unroll
            for (int n = 0; n < 4; ++n) o[n][j] *= scale;
        }

        // ---- O += P . V  (P from LDS as A-fragment, V^T rows contiguous) --
        v16bf pa = load_frag16(myP, 32);
#pragma unroll
        for (int n = 0; n < 4; ++n) {
            v16bf vb = load_frag16(Vt + (bh * D_SZ + n * 16) * P_SZ + s0, P_SZ);
            o[n] = wmma_bf16(pa, vb, o[n]);
        }
    }

    // ---- normalize and emit concatenated-head layout [B][P][H*D] ----
#pragma unroll
    for (int j = 0; j < 8; ++j) {
        const float inv = 1.0f / lrow[j];
        const int row = q0 + j + 8 * hh;
#pragma unroll
        for (int n = 0; n < 4; ++n) {
            const int col = head * D_SZ + n * 16 + r;
            AttnC[((size_t)batch * P_SZ + row) * M_SZ + col] = f2bf(o[n][j] * inv);
        }
    }
}

// ---------------------------------------------------------------------------
// Output projection: out[b][p][n] = AttnC[b][p][:] . Wout[:][n]   (f32 out)
//   A  : [B][P][M] bf16, Wt : [N][M] bf16 (transposed)
// ---------------------------------------------------------------------------
__global__ __launch_bounds__(256) void out_proj_kernel(
    const unsigned short* __restrict__ A,
    const unsigned short* __restrict__ Wt,
    float* __restrict__ Out) {
    const int mTile = blockIdx.x;                 // 0..15
    const int nTile = blockIdx.y;                 // 0..15 (64 cols each)
    const int batch = blockIdx.z;
    const int wave  = threadIdx.x >> 5;
    const int lane  = threadIdx.x & 31;
    const int q0 = mTile * 128 + wave * 16;
    const int n0 = nTile * 64;

    const unsigned short* Arow = A + ((size_t)batch * P_SZ + q0) * M_SZ;

    v8f acc[4] = {};
    for (int k = 0; k < M_SZ; k += 32) {
        v16bf a = load_frag16(Arow + k, M_SZ);
#pragma unroll
        for (int n = 0; n < 4; ++n) {
            v16bf b = load_frag16(Wt + (size_t)(n0 + n * 16) * M_SZ + k, M_SZ);
            acc[n] = wmma_bf16(a, b, acc[n]);
        }
    }

    const int r  = lane & 15;
    const int hh = lane >> 4;
#pragma unroll
    for (int n = 0; n < 4; ++n)
#pragma unroll
        for (int j = 0; j < 8; ++j)
            Out[((size_t)batch * P_SZ + q0 + j + 8 * hh) * M_SZ + n0 + n * 16 + r] =
                acc[n][j];
}

// ---------------------------------------------------------------------------
// Host-side orchestration
// ---------------------------------------------------------------------------
extern "C" void kernel_launch(void* const* d_in, const int* in_sizes, int n_in,
                              void* d_out, int out_size, void* d_ws, size_t ws_size,
                              hipStream_t stream) {
    const float* x   = (const float*)d_in[0];   // [B][P][M]
    const float* wq  = (const float*)d_in[1];   // [H][M][D]
    const float* wk  = (const float*)d_in[2];
    const float* wv  = (const float*)d_in[3];
    const float* wo  = (const float*)d_in[4];   // [M][M]
    float* out = (float*)d_out;                 // [B][P][M] f32

    // Workspace partition (bf16 stored as ushort)
    char* ws = (char*)d_ws;
    const size_t nX   = (size_t)B_SZ * P_SZ * M_SZ;          // 4M
    const size_t nWh  = (size_t)H_SZ * D_SZ * M_SZ;          // 1M
    const size_t nWo  = (size_t)M_SZ * M_SZ;                 // 1M
    const size_t nQKV = (size_t)B_SZ * H_SZ * P_SZ * D_SZ;   // 4M

    unsigned short* Xbf  = (unsigned short*)ws;                 ws += nX   * 2;
    unsigned short* WqT  = (unsigned short*)ws;                 ws += nWh  * 2;
    unsigned short* WkT  = (unsigned short*)ws;                 ws += nWh  * 2;
    unsigned short* WvT  = (unsigned short*)ws;                 ws += nWh  * 2;
    unsigned short* WoT  = (unsigned short*)ws;                 ws += nWo  * 2;
    unsigned short* Qb   = (unsigned short*)ws;                 ws += nQKV * 2;
    unsigned short* Kb   = (unsigned short*)ws;                 ws += nQKV * 2;
    unsigned short* Vt   = (unsigned short*)ws;                 ws += nQKV * 2;
    unsigned short* AtC  = (unsigned short*)ws;                 ws += nX   * 2;

    // 1) casts / transposes
    cast_f32_bf16_kernel<<<(int)((nX + 255) / 256), 256, 0, stream>>>(x, Xbf, (int)nX);
    transpose_whead_kernel<<<(int)(nWh / 256), 256, 0, stream>>>(wq, WqT);
    transpose_whead_kernel<<<(int)(nWh / 256), 256, 0, stream>>>(wk, WkT);
    transpose_whead_kernel<<<(int)(nWh / 256), 256, 0, stream>>>(wv, WvT);
    transpose_wout_kernel<<<(int)(nWo / 256), 256, 0, stream>>>(wo, WoT);

    // 2) QKV projections (V stored transposed for the PV matmul)
    dim3 gProj(P_SZ / 128, H_SZ, B_SZ);
    qkv_proj_kernel<<<gProj, 256, 0, stream>>>(Xbf, WqT, Qb, 0);
    qkv_proj_kernel<<<gProj, 256, 0, stream>>>(Xbf, WkT, Kb, 0);
    qkv_proj_kernel<<<gProj, 256, 0, stream>>>(Xbf, WvT, Vt, 1);

    // 3) causal flash attention -> concatenated heads
    flash_attn_kernel<<<gProj, 256, 0, stream>>>(Qb, Kb, Vt, AtC);

    // 4) output projection (f32 result)
    dim3 gOut(P_SZ / 128, M_SZ / 64, B_SZ);
    out_proj_kernel<<<gOut, 256, 0, stream>>>(AtC, WoT, out);
}
